// Head_11733850653001
// MI455X (gfx1250) — matrix operand order repaired and checked by reading, not compiled, not run on hardware
//
#include <hip/hip_runtime.h>
#include <hip/hip_bf16.h>
#include <stdint.h>

// Problem dims (compile-time)
#define Bn 4
#define Tn 4096
#define En 1024
#define Hn 128
#define BT (Bn * Tn)

typedef __attribute__((ext_vector_type(16))) __bf16 v16bf;
typedef __attribute__((ext_vector_type(8)))  __bf16 v8bf;
typedef __attribute__((ext_vector_type(8)))  float  v8f;
typedef __attribute__((ext_vector_type(4)))  unsigned int v4u;
typedef __attribute__((ext_vector_type(8)))  int    v8i;
typedef __attribute__((ext_vector_type(4)))  int    v4i;

static __device__ inline v8f wmma_bf16(v16bf a, v16bf b, v8f c) {
  // (neg_a, A, neg_b, B, c_mod, C, reuse_a, reuse_b)
  return __builtin_amdgcn_wmma_f32_16x16x32_bf16(false, a, false, b, (short)0, c,
                                                 false, false);
}

// Build 16-bit A-fragment from two contiguous 8-element runs
// (k order used here, matched in B: lane<16 -> M=lane, K=[0..15]; lane>=16 -> M=lane-16, K=[16..31])
static __device__ inline v16bf make_afrag(v8bf lo, v8bf hi) {
  return __builtin_shufflevector(lo, hi, 0, 1, 2, 3, 4, 5, 6, 7,
                                         8, 9, 10, 11, 12, 13, 14, 15);
}

static __device__ inline float half_reduce_max(float x) {
#pragma unroll
  for (int m = 1; m <= 8; m <<= 1) x = fmaxf(x, __shfl_xor(x, m, 32));
  return x;
}
static __device__ inline float half_reduce_sum(float x) {
#pragma unroll
  for (int m = 1; m <= 8; m <<= 1) x += __shfl_xor(x, m, 32);
  return x;
}

// ---------------------------------------------------------------------------
// TDM: async 2D tile load Global -> LDS (Tensor Data Mover, TENSORcnt-tracked).
// Dimensions in 8-byte units (D#.data_size = 3). Row-major tile:
//   tile_d1 rows of tile_d0 qwords, source row stride d0_stride_qw.
// ---------------------------------------------------------------------------
static __device__ inline void tdm_load_2d(unsigned lds_off, const void* gaddr,
                                          unsigned tile_d0_qw, unsigned tile_d1,
                                          unsigned tensor_d0_qw,
                                          unsigned tensor_d1,
                                          unsigned d0_stride_qw) {
  const unsigned long long ga = (unsigned long long)(size_t)gaddr;
  v4u g0;
  g0[0] = 1u;                                   // count=1, is_restore=0
  g0[1] = lds_off;                              // D#.lds_addr (bytes)
  g0[2] = (unsigned)(ga & 0xFFFFFFFFu);         // global_addr[31:0]
  g0[3] = (unsigned)((ga >> 32) & 0x1FFFFFFu)   // global_addr[56:32]
          | (2u << 30);                         // type=2 ("image")
  v8i g1;
  g1[0] = (int)(3u << 16);                      // workgroup_mask=0, data_size=8B
  g1[1] = (int)((tensor_d0_qw & 0xFFFFu) << 16);              // dim0[15:0]
  g1[2] = (int)(((tensor_d0_qw >> 16) & 0xFFFFu) |            // dim0[31:16]
                ((tensor_d1 & 0xFFFFu) << 16));               // dim1[15:0]
  g1[3] = (int)(((tensor_d1 >> 16) & 0xFFFFu) |               // dim1[31:16]
                ((tile_d0_qw & 0xFFFFu) << 16));              // tile_dim0
  g1[4] = (int)(tile_d1 & 0xFFFFu);             // tile_dim1 (tile_dim2=0)
  g1[5] = (int)d0_stride_qw;                    // tensor_dim0_stride[31:0]
  g1[6] = 0;                                    // stride hi / dim1_stride lo
  g1[7] = 0;
  const v4i gz = {0, 0, 0, 0};
#if __clang_major__ >= 23
  const v8i gz8 = {0, 0, 0, 0, 0, 0, 0, 0};
  __builtin_amdgcn_tensor_load_to_lds(g0, g1, gz, gz, gz8, 0);
#else
  __builtin_amdgcn_tensor_load_to_lds(g0, g1, gz, gz, 0);
#endif
}

// ---------------------------------------------------------------------------
// Transpose + convert weights: W f32 [E][H] -> Wt bf16 [H][E]
// ---------------------------------------------------------------------------
__global__ void wt_transpose_kernel(const float* __restrict__ W,
                                    __bf16* __restrict__ Wt) {
  const int idx = blockIdx.x * blockDim.x + threadIdx.x; // over H*E
  const int e = idx & (En - 1);
  const int h = idx >> 10; // log2(En)
  Wt[(size_t)h * En + e] = (__bf16)W[(size_t)e * Hn + h];
}

// ---------------------------------------------------------------------------
// Projection: Y[BT,H] = X[BT,E] @ W[E,H]; one wave per 16-row M tile, N=128.
// Per-chunk Wt tile (128 x 32 bf16 = 8KB) is TDM-staged into double-buffered
// LDS and shared by all 8 waves (8x L2 traffic reduction, LDS-fed WMMA).
// ---------------------------------------------------------------------------
__global__ void proj_kernel(const float* __restrict__ X,
                            const __bf16* __restrict__ Wt,
                            __bf16* __restrict__ Yrm,
                            __bf16* __restrict__ Ytr, float scale) {
  __shared__ __align__(64) __bf16 wbuf[2][128 * 32]; // [h][e-local]

  const int wave = threadIdx.x >> 5;
  const int lane = threadIdx.x & 31;
  const int hg   = lane >> 4; // half-wave group
  const int l16  = lane & 15;
  const int row0 = blockIdx.x * 128 + wave * 16;

  v8f acc[8];
  const v8f vz = {0.f, 0.f, 0.f, 0.f, 0.f, 0.f, 0.f, 0.f};
#pragma unroll
  for (int nt = 0; nt < 8; ++nt) acc[nt] = vz;

  const int nchunks = En / 32;
  // Prologue: stage chunk 0 (Wt[0..127][0..31]: 128 rows x 8 qw, stride 256 qw)
  if (wave == 0)
    tdm_load_2d((unsigned)(size_t)&wbuf[0][0], Wt, 8u, 128u, 8u, 128u,
                (unsigned)(En * 2 / 8));

  const int arow = row0 + l16;
  for (int i = 0; i < nchunks; ++i) {
    const int e0  = i * 32;
    const int cur = i & 1;
    if (wave == 0) {
      if (i + 1 < nchunks) {
        tdm_load_2d((unsigned)(size_t)&wbuf[cur ^ 1][0], Wt + (e0 + 32), 8u,
                    128u, 8u, 128u, (unsigned)(En * 2 / 8));
        __builtin_amdgcn_s_wait_tensorcnt(1); // in-order: chunk i has landed
      } else {
        __builtin_amdgcn_s_wait_tensorcnt(0);
      }
    }
    __syncthreads();

    // A fragment: two 32B f32 loads from X, convert to bf16
    const float* ap = X + (size_t)arow * En + e0 + hg * 16;
    if (i + 1 < nchunks) __builtin_prefetch(ap + 32, 0, 0);
    v8f flo = *(const v8f*)ap;
    v8f fhi = *(const v8f*)(ap + 8);
    v16bf a;
#pragma unroll
    for (int j = 0; j < 8; ++j) {
      a[j]     = (__bf16)flo[j];
      a[8 + j] = (__bf16)fhi[j];
    }
#pragma unroll
    for (int nt = 0; nt < 8; ++nt) {
      // B fragment from LDS: row h = nt*16+l16, 16 contiguous K values
      const __bf16* bp = &wbuf[cur][(nt * 16 + l16) * 32 + hg * 16];
      v16bf b = *(const v16bf*)bp;
      acc[nt] = wmma_bf16(a, b, acc[nt]);
    }
    __syncthreads(); // compute done before buf[cur] is overwritten (i+2)
  }

  // C layout: lane<16 -> N=lane, M=r; lane>=16 -> N=lane-16, M=8+r
#pragma unroll
  for (int nt = 0; nt < 8; ++nt) {
#pragma unroll
    for (int r = 0; r < 8; ++r) {
      const int row = row0 + hg * 8 + r;
      const int h   = nt * 16 + l16;
      const float y = acc[nt][r] * scale;
      if (Yrm) Yrm[(size_t)row * Hn + h] = (__bf16)y;
      if (Ytr) {
        const int bb = row >> 12;      // / Tn
        const int tt = row & (Tn - 1); // % Tn
        Ytr[((size_t)bb * Hn + h) * Tn + tt] = (__bf16)y;
      }
    }
  }
}

// ---------------------------------------------------------------------------
// Causal flash attention. One wave = 16 query rows; key tiles of 32 shared by
// all 8 waves of the block and TDM-staged into double-buffered LDS.
// qp: bf16 [B][T][H] (pre-scaled by H^-0.5), kp: bf16 [B][T][H],
// vT: bf16 [B][H][T], out: f32 [B][T][H].
// ---------------------------------------------------------------------------
__global__ void attn_kernel(const __bf16* __restrict__ qp,
                            const __bf16* __restrict__ kp,
                            const __bf16* __restrict__ vT,
                            float* __restrict__ out) {
  __shared__ __align__(64) __bf16 kbuf[2][32 * 128]; // [key-local][h]
  __shared__ __align__(64) __bf16 vbuf[2][128 * 32]; // [h][key-local]
  __shared__ __align__(64) __bf16 pbuf[8][16 * 32];  // per-wave P tile

  const int wave = threadIdx.x >> 5;
  const int lane = threadIdx.x & 31;
  const int hg   = lane >> 4;
  const int l16  = lane & 15;
  const int tiles_per_b = Tn / 128;
  const int b      = blockIdx.x / tiles_per_b;
  const int q0_blk = (blockIdx.x % tiles_per_b) * 128;
  const int q0     = q0_blk + wave * 16;

  // Q fragments held for the whole KV loop (4 h-chunks of 32)
  v16bf qA[4];
  {
    const __bf16* qb = qp + ((size_t)b * Tn + q0 + l16) * Hn;
#pragma unroll
    for (int hc = 0; hc < 4; ++hc) {
      const __bf16* p = qb + hc * 32 + hg * 16;
      qA[hc] = make_afrag(*(const v8bf*)p, *(const v8bf*)(p + 8));
    }
  }

  const v8f vz = {0.f, 0.f, 0.f, 0.f, 0.f, 0.f, 0.f, 0.f};
  v8f o[8];
#pragma unroll
  for (int nt = 0; nt < 8; ++nt) o[nt] = vz;
  float mrow[8], lrow[8];
#pragma unroll
  for (int r = 0; r < 8; ++r) { mrow[r] = -1e30f; lrow[r] = 0.f; }

  const int kend   = q0 + 16;                  // this wave's causal bound
  const int ntiles = (q0_blk + 128) / 32;      // block-uniform trip count

  // Prologue: stage tile 0 (K: 32x128 contiguous; V^T: 128 rows x 8 qw strided)
  if (wave == 0) {
    const __bf16* kg = kp + ((size_t)b * Tn) * Hn;
    const __bf16* vg = vT + ((size_t)b * Hn) * Tn;
    tdm_load_2d((unsigned)(size_t)&kbuf[0][0], kg, 32u, 32u, 32u, 32u, 32u);
    tdm_load_2d((unsigned)(size_t)&vbuf[0][0], vg, 8u, 128u, 8u, 128u,
                (unsigned)(Tn * 2 / 8));
  }

  for (int i = 0; i < ntiles; ++i) {
    const int kv0 = i * 32;
    const int cur = i & 1;
    if (wave == 0) {
      if (i + 1 < ntiles) {
        const int kn = kv0 + 32;
        const __bf16* kg = kp + ((size_t)b * Tn + kn) * Hn;
        const __bf16* vg = vT + ((size_t)b * Hn) * Tn + kn;
        tdm_load_2d((unsigned)(size_t)&kbuf[cur ^ 1][0], kg, 32u, 32u, 32u, 32u,
                    32u);
        tdm_load_2d((unsigned)(size_t)&vbuf[cur ^ 1][0], vg, 8u, 128u, 8u, 128u,
                    (unsigned)(Tn * 2 / 8));
        __builtin_amdgcn_s_wait_tensorcnt(2); // in-order: tile i's pair landed
      } else {
        __builtin_amdgcn_s_wait_tensorcnt(0);
      }
    }
    __syncthreads();

    if (kv0 < kend) { // wave-uniform: this wave still has unmasked keys
      // S tiles (16x16 each) from LDS K tile
      v8f s0 = vz, s1 = vz;
#pragma unroll
      for (int hc = 0; hc < 4; ++hc) {
        const __bf16* kb0 = &kbuf[cur][l16 * 128 + hc * 32 + hg * 16];
        const __bf16* kb1 = kb0 + 16 * 128;
        s0 = wmma_bf16(qA[hc], *(const v16bf*)kb0, s0);
        s1 = wmma_bf16(qA[hc], *(const v16bf*)kb1, s1);
      }

      // causal mask + online softmax (rows distributed over half-wave groups)
      const int kcol0 = kv0 + l16;
      const int kcol1 = kv0 + 16 + l16;
#pragma unroll
      for (int r = 0; r < 8; ++r) {
        const int q = q0 + hg * 8 + r;
        const float a0 = (kcol0 <= q) ? s0[r] : -1e30f;
        const float a1 = (kcol1 <= q) ? s1[r] : -1e30f;
        const float mx = half_reduce_max(fmaxf(a0, a1));
        const float mn = fmaxf(mrow[r], mx);
        const float ef = __expf(mrow[r] - mn);
        const float p0 = __expf(a0 - mn);
        const float p1 = __expf(a1 - mn);
        lrow[r] = lrow[r] * ef + half_reduce_sum(p0 + p1);
        mrow[r] = mn;
        s0[r] = p0;
        s1[r] = p1;
#pragma unroll
        for (int nt = 0; nt < 8; ++nt) o[nt][r] *= ef;
      }

      // C-layout f32 -> A-layout bf16 via per-wave LDS tile
      __bf16* pb = &pbuf[wave][0];
#pragma unroll
      for (int r = 0; r < 8; ++r) {
        const int rl = hg * 8 + r;
        pb[rl * 32 + l16]      = (__bf16)s0[r];
        pb[rl * 32 + 16 + l16] = (__bf16)s1[r];
      }
      asm volatile("s_wait_dscnt 0" ::: "memory"); // DS RAW fence within wave

      const __bf16* pp = pb + l16 * 32 + hg * 16;
      const v16bf pA = make_afrag(*(const v8bf*)pp, *(const v8bf*)(pp + 8));

      // O += P @ V : B fragments from LDS V^T tile
#pragma unroll
      for (int nt = 0; nt < 8; ++nt) {
        const __bf16* vb = &vbuf[cur][(nt * 16 + l16) * 32 + hg * 16];
        o[nt] = wmma_bf16(pA, *(const v16bf*)vb, o[nt]);
      }
    }
    __syncthreads(); // all waves done with buf[cur] before tile i+2 overwrite
  }

  // epilogue: divide by softmax denominator, store f32
  float* ob = out + ((size_t)b * Tn + q0) * Hn;
#pragma unroll
  for (int nt = 0; nt < 8; ++nt) {
#pragma unroll
    for (int r = 0; r < 8; ++r) {
      const float inv_l = 1.0f / lrow[r];
      ob[(size_t)(hg * 8 + r) * Hn + nt * 16 + l16] = o[nt][r] * inv_l;
    }
  }
}

// ---------------------------------------------------------------------------
// Launch: inputs are (k, q, v, Wk, Wq, Wv), all f32. Output f32 [B,T,H].
// ---------------------------------------------------------------------------
extern "C" void kernel_launch(void* const* d_in, const int* in_sizes, int n_in,
                              void* d_out, int out_size, void* d_ws,
                              size_t ws_size, hipStream_t stream) {
  const float* k  = (const float*)d_in[0];
  const float* q  = (const float*)d_in[1];
  const float* v  = (const float*)d_in[2];
  const float* Wk = (const float*)d_in[3];
  const float* Wq = (const float*)d_in[4];
  const float* Wv = (const float*)d_in[5];
  float* out = (float*)d_out;

  uint8_t* w = (uint8_t*)d_ws;
  __bf16* qp  = (__bf16*)w; w += (size_t)BT * Hn * sizeof(__bf16);
  __bf16* kp  = (__bf16*)w; w += (size_t)BT * Hn * sizeof(__bf16);
  __bf16* vT  = (__bf16*)w; w += (size_t)BT * Hn * sizeof(__bf16);
  __bf16* Wtq = (__bf16*)w; w += (size_t)Hn * En * sizeof(__bf16);
  __bf16* Wtk = (__bf16*)w; w += (size_t)Hn * En * sizeof(__bf16);
  __bf16* Wtv = (__bf16*)w; w += (size_t)Hn * En * sizeof(__bf16);

  const dim3 blk(256);

  wt_transpose_kernel<<<(Hn * En) / 256, blk, 0, stream>>>(Wq, Wtq);
  wt_transpose_kernel<<<(Hn * En) / 256, blk, 0, stream>>>(Wk, Wtk);
  wt_transpose_kernel<<<(Hn * En) / 256, blk, 0, stream>>>(Wv, Wtv);

  const float scale = 0.08838834764831845f; // 128^-0.5 folded into qp
  proj_kernel<<<BT / 128, blk, 0, stream>>>(q, Wtq, qp, (__bf16*)nullptr, scale);
  proj_kernel<<<BT / 128, blk, 0, stream>>>(k, Wtk, kp, (__bf16*)nullptr, 1.0f);
  proj_kernel<<<BT / 128, blk, 0, stream>>>(v, Wtv, (__bf16*)nullptr, vT, 1.0f);

  attn_kernel<<<Bn * (Tn / 128), blk, 0, stream>>>(qp, kp, vT, out);
}
